// ODConv2D_42021960024321
// MI455X (gfx1250) — matrix operand order repaired and checked by reading, not compiled, not run on hardware
//
#include <hip/hip_runtime.h>

#define BN   16
#define CIN  128
#define COUT 256
#define KS   3
#define HH   56
#define WW   56
#define HW   (HH * WW)        // 3136
#define TAPS 9
#define RTOT (COUT * CIN * TAPS)   // 294912

typedef __attribute__((ext_vector_type(2)))  float    v2f;
typedef __attribute__((ext_vector_type(4)))  float    v4f;
typedef __attribute__((ext_vector_type(8)))  float    v8f;
typedef __attribute__((ext_vector_type(8)))  _Float16 v8h;
typedef __attribute__((ext_vector_type(16))) _Float16 v16h;
typedef __attribute__((ext_vector_type(4)))  unsigned int v4u;
typedef __attribute__((ext_vector_type(4)))  int      v4i;
typedef __attribute__((ext_vector_type(8)))  int      v8i;

union V16H { v16h v; v8h h[2]; };

__device__ __forceinline__ float sigmoidf_fast(float x) {
  return 1.0f / (1.0f + __expf(-x));
}

// ---------------------------------------------------------------------------
// TDM: DMA a 2D f16 tile [COUT rows x 32 halves], row stride CIN halves,
// from global into LDS at byte offset lds_off. D# per CDNA5 ISA:
//   group0: count=1 | lds_addr | global_addr[56:0] | type=2
//   group1: data_size=1(2B) | tensor_dim0=CIN | tensor_dim1=COUT |
//           tile_dim0=32 | tile_dim1=COUT | tensor_dim0_stride=CIN
// ---------------------------------------------------------------------------
__device__ __forceinline__ void tdm_load_A(const _Float16* gsrc,
                                           unsigned lds_off) {
  unsigned long long ga = (unsigned long long)(uintptr_t)gsrc;
  v4u g0 = { 1u,                                   // count=1, user mode
             lds_off,                              // lds_addr (bytes)
             (unsigned)(ga & 0xFFFFFFFFu),         // global_addr[31:0]
             (unsigned)((ga >> 32) & 0x01FFFFFFu)  // global_addr[56:32]
                 | 0x80000000u };                  // type=2 ("image")
  v8i g1 = { 0x10000,                              // data_size=1 -> 2 bytes
             (int)(CIN << 16),                     // tensor_dim0 lo16
             (int)(COUT << 16),                    // dim0 hi=0 | tensor_dim1 lo16
             (int)(32 << 16),                      // tensor_dim1 hi=0 | tile_dim0
             COUT,                                 // tile_dim1 | tile_dim2=0
             CIN,                                  // tensor_dim0_stride lo32
             0, 0 };                               // stride0 hi | stride1
  v4i z4 = { 0, 0, 0, 0 };
#if defined(__clang_major__) && (__clang_major__ >= 23)
  v8i z8 = { 0, 0, 0, 0, 0, 0, 0, 0 };
  __builtin_amdgcn_tensor_load_to_lds(g0, g1, z4, z4, z8, 0);
#else
  __builtin_amdgcn_tensor_load_to_lds(g0, g1, z4, z4, 0);
#endif
}

// ---------------------------------------------------------------------------
// Kernel A: global average pool  x[B,CIN,H,W] -> ctx[B*CIN]
// ---------------------------------------------------------------------------
__global__ void __launch_bounds__(128)
gap_kernel(const float* __restrict__ x, float* __restrict__ ctx) {
  __shared__ float red[128];
  const int bc = blockIdx.x;                 // 0..B*CIN-1
  const float* p = x + (size_t)bc * HW;
  float s = 0.0f;
  for (int i = threadIdx.x; i < HW; i += 128) s += p[i];
  red[threadIdx.x] = s;
  __syncthreads();
  for (int off = 64; off > 0; off >>= 1) {
    if (threadIdx.x < off) red[threadIdx.x] += red[threadIdx.x + off];
    __syncthreads();
  }
  if (threadIdx.x == 0) ctx[bc] = red[0] * (1.0f / (float)HW);
}

// ---------------------------------------------------------------------------
// Kernel B: small attention heads (spatial 9, in 128, out 256) per sample
// ---------------------------------------------------------------------------
__global__ void __launch_bounds__(512)
attn_small_kernel(const float* __restrict__ ctx,
                  const float* __restrict__ fsw, const float* __restrict__ fsb,
                  const float* __restrict__ fiw, const float* __restrict__ fib,
                  const float* __restrict__ fow, const float* __restrict__ fob,
                  float* __restrict__ a_sp, float* __restrict__ a_in,
                  float* __restrict__ a_out) {
  __shared__ float c[CIN];
  const int b = blockIdx.x;
  if (threadIdx.x < CIN) c[threadIdx.x] = ctx[b * CIN + threadIdx.x];
  __syncthreads();
  const int t = threadIdx.x;
  const float* wrow;
  float bias;
  float* dst;
  if (t < TAPS) {
    wrow = fsw + t * CIN; bias = fsb[t]; dst = a_sp + b * TAPS + t;
  } else if (t < TAPS + CIN) {
    const int r = t - TAPS;
    wrow = fiw + r * CIN; bias = fib[r]; dst = a_in + b * CIN + r;
  } else if (t < TAPS + CIN + COUT) {
    const int r = t - TAPS - CIN;
    wrow = fow + r * CIN; bias = fob[r]; dst = a_out + b * COUT + r;
  } else {
    return;
  }
  float acc = bias;
  for (int k = 0; k < CIN; ++k) acc += c[k] * wrow[k];
  *dst = sigmoidf_fast(acc);
}

// ---------------------------------------------------------------------------
// Kernel C: dynamic-weight generation (memory-bound GEMM over fc_kernel_w).
//   D[16 rows, 16 batches] = W_tile[16,128] x ctx^T[128,16]  via
//   V_WMMA_F32_16X16X4_F32, K=128 in 32 steps; fused sigmoid + 4-way attn
//   broadcast; output f16 in layout [b][tap][o][i] for the conv A-matrix.
// Block = 128 threads = 4 wave32, one 16-row tile per wave.
// ---------------------------------------------------------------------------
__global__ void __launch_bounds__(128)
dyn_weight_kernel(const float* __restrict__ ctx,
                  const float* __restrict__ fkw,   // [RTOT][CIN]
                  const float* __restrict__ fkb,   // [RTOT]
                  const float* __restrict__ base,  // [COUT][CIN][3][3] flat
                  const float* __restrict__ a_sp,  // [B][9]
                  const float* __restrict__ a_in,  // [B][CIN]
                  const float* __restrict__ a_out, // [B][COUT]
                  _Float16* __restrict__ dyn) {    // [B][9][COUT][CIN]
  __shared__ float ctx_s[BN * CIN];        // 8 KB
  __shared__ float At[4][16 * CIN];        // 32 KB
  const int tid  = threadIdx.x;
  const int wave = tid >> 5;
  const int lane = tid & 31;
  const int half = lane >> 4;              // K-half selector
  const int lm   = lane & 15;

  // stage context [16][128] into LDS (coalesced)
  for (int j = tid; j < BN * CIN; j += 128) ctx_s[j] = ctx[j];

  // stage this wave's 16x128 f32 tile of fc_kernel_w (coalesced v4f rows)
  const int r0 = blockIdx.x * 64 + wave * 16;
  for (int rr = 0; rr < 16; ++rr) {
    v4f val = *(const v4f*)(fkw + (size_t)(r0 + rr) * CIN + lane * 4);
    *(v4f*)(&At[wave][rr * CIN + lane * 4]) = val;
  }
  // hint-prefetch the next tile region of the 151MB stream
  __builtin_prefetch(fkw + (size_t)(r0 + 64) * CIN + lane * 4, 0, 1);
  __syncthreads();

  // WMMA f32 16x16x4 accumulate across K=128
  v8f acc = {};
  const float* arow = &At[wave][lm * CIN + 2 * half];  // A: M=lm, K base by half
  const float* brow = &ctx_s[lm * CIN + 2 * half];     // B[k][n]=ctx[n][k]: N=lm
#pragma unroll
  for (int s = 0; s < 32; ++s) {
    v2f a   = *(const v2f*)(arow + 4 * s);
    v2f bfr = *(const v2f*)(brow + 4 * s);
    acc = __builtin_amdgcn_wmma_f32_16x16x4_f32(
        false, a, false, bfr, (short)0, acc, false, false);
  }

  // epilogue: lane holds batch b = lm, rows m = v + 8*half
  const int b = lm;
#pragma unroll
  for (int v = 0; v < 8; ++v) {
    const int m = v + 8 * half;
    const int r = r0 + m;
    const int o   = r / (CIN * TAPS);
    const int rem = r - o * (CIN * TAPS);
    const int i   = rem / TAPS;
    const int tap = rem - i * TAPS;
    const float t   = acc[v] + fkb[r];
    const float val = base[r] * a_out[b * COUT + o] * a_in[b * CIN + i] *
                      a_sp[b * TAPS + tap] * sigmoidf_fast(t);
    dyn[(((size_t)b * TAPS + tap) * COUT + o) * CIN + i] = (_Float16)val;
  }
}

// ---------------------------------------------------------------------------
// Kernel D: per-sample implicit-GEMM conv with V_WMMA_F32_16X16X32_F16.
// Block = 256 threads = 8 wave32. Tile: M=256 (all COUT) x N=64 pixels.
// Flattened K-loop: 36 stages (4 CIN-chunks x 9 taps), K=32 per WMMA step.
// A tile (16KB) DMA'd by the Tensor Data Mover into double-buffered LDS,
// pipelined one stage ahead (s_wait_tensorcnt(1) retires the current stage's
// DMA while the next is in flight). B tile im2col'd f32->f16 by the VALU.
// ---------------------------------------------------------------------------
__global__ void __launch_bounds__(256)
conv_kernel(const float* __restrict__ x,        // [B][CIN][56][56]
            const _Float16* __restrict__ dyn,   // [B][9][COUT][CIN]
            const float* __restrict__ bias,     // [COUT]
            float* __restrict__ out) {          // [B][COUT][56][56]
  __shared__ _Float16 Als[2][COUT * 32];  // 2 x 16 KB, [o][i] rows of 32
  __shared__ _Float16 Bls[64 * 32];       // 4 KB,  [p][i] rows of 32 halves
  const int tid  = threadIdx.x;
  const int wave = tid >> 5;              // 0..7
  const int lane = tid & 31;
  const int half = lane >> 4;
  const int lm   = lane & 15;
  const int b    = blockIdx.y;
  const int p0   = blockIdx.x * 64;

  v8f acc[2][4];
  const v8f vzero = {};
#pragma unroll
  for (int mt = 0; mt < 2; ++mt)
#pragma unroll
    for (int nt = 0; nt < 4; ++nt) acc[mt][nt] = vzero;

  // B-staging coords: this thread fills pixel (tid&63), channel group (tid>>6)
  const int pl = tid & 63;
  const int cg = tid >> 6;              // 0..3 -> 8 channels each
  const int p  = p0 + pl;
  const int py = p / WW;
  const int px = p - py * WW;

  const _Float16* dynb = dyn + (size_t)b * TAPS * COUT * CIN;

  // prologue: TDM for stage 0 (tap 0, i0 0) into buffer 0
  if (wave == 0) {
    tdm_load_A(dynb, (unsigned)(uintptr_t)&Als[0][0]);
  }

  for (int s = 0; s < 36; ++s) {
    const int ic  = s / TAPS;
    const int tap = s - ic * TAPS;
    const int i0  = ic * 32;
    const int kh  = tap / 3 - 1;
    const int kw  = tap % 3 - 1;

    // stage B: x values for pixel pl, channels i0+8cg..+7, f32->f16 packed
    {
      const int yy = py + kh;
      const int xx = px + kw;
      const bool ok = (yy >= 0) && (yy < HH) && (xx >= 0) && (xx < WW);
      const float* xs =
          x + ((size_t)b * CIN + (i0 + cg * 8)) * HW + yy * WW + xx;
      v8h pk;
#pragma unroll
      for (int j = 0; j < 8; ++j) {
        float vx = ok ? xs[j * HW] : 0.0f;
        pk[j] = (_Float16)vx;
      }
      *(v8h*)(&Bls[pl * 32 + cg * 8]) = pk;
    }

    // pipeline: issue next stage's TDM, then retire current stage's DMA
    if (wave == 0) {
      if (s + 1 < 36) {
        const int sn   = s + 1;
        const int icn  = sn / TAPS;
        const int tapn = sn - icn * TAPS;
        tdm_load_A(dynb + (size_t)tapn * COUT * CIN + icn * 32,
                   (unsigned)(uintptr_t)&Als[sn & 1][0]);
        __builtin_amdgcn_s_wait_tensorcnt(1);   // stage s done, s+1 in flight
      } else {
        __builtin_amdgcn_s_wait_tensorcnt(0);
      }
    }
    __syncthreads();

    // B fragments: lane = col lm, K 8h..8h+7 then 16+8h..16+8h+7 (matches A)
    const _Float16* Acur = &Als[s & 1][0];
    V16H bfrag[4];
#pragma unroll
    for (int nt = 0; nt < 4; ++nt) {
      const _Float16* bp = &Bls[(nt * 16 + lm) * 32 + half * 8];
      bfrag[nt].h[0] = *(const v8h*)(bp);
      bfrag[nt].h[1] = *(const v8h*)(bp + 16);
    }
    // A fragments + 8 WMMA per wave per K-step
#pragma unroll
    for (int mt = 0; mt < 2; ++mt) {
      const int m = wave * 32 + mt * 16 + lm;
      const _Float16* ap = Acur + m * 32;
      V16H afrag;
      afrag.h[0] = *(const v8h*)(ap + half * 8);        // K 0..7 / 8..15
      afrag.h[1] = *(const v8h*)(ap + 16 + half * 8);   // K 16..23 / 24..31
#pragma unroll
      for (int nt = 0; nt < 4; ++nt) {
        acc[mt][nt] = __builtin_amdgcn_wmma_f32_16x16x32_f16(
            false, afrag.v, false, bfrag[nt].v, (short)0, acc[mt][nt],
            false, false);
      }
    }
    __syncthreads();   // all reads of Bls / Als[s&1] done before next stage
  }

  // epilogue: D layout lane=N col, vgpr v -> M = v + 8*half
#pragma unroll
  for (int mt = 0; mt < 2; ++mt) {
#pragma unroll
    for (int nt = 0; nt < 4; ++nt) {
      const int n = p0 + nt * 16 + lm;
#pragma unroll
      for (int v = 0; v < 8; ++v) {
        const int m = wave * 32 + mt * 16 + v + 8 * half;
        out[((size_t)b * COUT + m) * HW + n] = acc[mt][nt][v] + bias[m];
      }
    }
  }
}

// ---------------------------------------------------------------------------
extern "C" void kernel_launch(void* const* d_in, const int* in_sizes, int n_in,
                              void* d_out, int out_size, void* d_ws,
                              size_t ws_size, hipStream_t stream) {
  const float* x    = (const float*)d_in[0];
  const float* base = (const float*)d_in[1];
  const float* bias = (const float*)d_in[2];
  const float* fsw  = (const float*)d_in[3];
  const float* fsb  = (const float*)d_in[4];
  const float* fiw  = (const float*)d_in[5];
  const float* fib  = (const float*)d_in[6];
  const float* fow  = (const float*)d_in[7];
  const float* fob  = (const float*)d_in[8];
  const float* fkw  = (const float*)d_in[9];
  const float* fkb  = (const float*)d_in[10];
  float* out = (float*)d_out;

  // workspace layout
  float* ctx   = (float*)d_ws;               // 2048 f32
  float* a_sp  = ctx + BN * CIN;             // 144 f32
  float* a_in  = a_sp + BN * TAPS;           // 2048 f32
  float* a_out = a_in + BN * CIN;            // 4096 f32
  _Float16* dyn = (_Float16*)((char*)d_ws + 65536);  // 9.44 MB f16

  gap_kernel<<<dim3(BN * CIN), dim3(128), 0, stream>>>(x, ctx);
  attn_small_kernel<<<dim3(BN), dim3(512), 0, stream>>>(
      ctx, fsw, fsb, fiw, fib, fow, fob, a_sp, a_in, a_out);
  dyn_weight_kernel<<<dim3(RTOT / 64), dim3(128), 0, stream>>>(
      ctx, fkw, fkb, base, a_sp, a_in, a_out, dyn);
  conv_kernel<<<dim3(HW / 64, BN), dim3(256), 0, stream>>>(x, dyn, bias, out);
}
